// TrueNeuron_49160195670184
// MI455X (gfx1250) — matrix operand order repaired and checked by reading, not compiled, not run on hardware
//
#include <hip/hip_runtime.h>

// ============================================================================
// Shapes (compile-time, from the reference): B=8, C=8, O=16, S=512, F=512, E=512
// ============================================================================
#define B_ 8
#define C_ 8
#define O_ 16
#define S_ 512
#define F_ 512
#define E_ 512
#define ND_INV 0.5f   // 1 / norm_decay(2.0)

typedef __bf16 bf16;
typedef __attribute__((ext_vector_type(2)))  __bf16 v2bf;
typedef __attribute__((ext_vector_type(4)))  __bf16 v4bf;
typedef __attribute__((ext_vector_type(8)))  __bf16 v8bf;
typedef __attribute__((ext_vector_type(16))) __bf16 v16bf;
typedef __attribute__((ext_vector_type(8)))  float  v8f;

// B-operand fragment-major LDS layout:
//   element (k,e) of a 32xN tile -> Bsh[(e>>4)*BF_NT + ((k>=16)*16 + (e&15))*BF_LANE + (k&15)]
//   so WMMA lane L (= K-half*16 + n) reads its 16 fragment bf16s contiguously.
#define BF_LANE 24   // 16 payload + 8 pad bf16 (48B: 12-bank stride)
#define BF_NT   392  // 16*24 + 8 pad (784B per 16-col group)

// ---------------------------------------------------------------------------
// WMMA helpers (CDNA5 wave32, layouts per CDNA5 ISA §7.12.2)
// ---------------------------------------------------------------------------
__device__ __forceinline__ v8f wmma_bf16(v16bf a, v16bf b, v8f c) {
  return __builtin_amdgcn_wmma_f32_16x16x32_bf16(
      false, a, false, b, (short)0, c, false, false);
}

__device__ __forceinline__ v16bf frag_from(const bf16* p0, const bf16* p1) {
  v8bf lo = *(const v8bf*)p0;   // ds_load_b128
  v8bf hi = *(const v8bf*)p1;   // ds_load_b128
  return __builtin_shufflevector(lo, hi, 0, 1, 2, 3, 4, 5, 6, 7,
                                 8, 9, 10, 11, 12, 13, 14, 15);
}

__device__ __forceinline__ v16bf load_a_frag(const bf16* A, int ldm) {
  const int lane = threadIdx.x & 31;
  const int m    = lane & 15;
  const int ko   = (lane >> 4) << 4;   // 0 or 16
  const bf16* p  = A + m * ldm + ko;
  return frag_from(p, p + 8);
}

__device__ __forceinline__ v16bf load_b_frag(const bf16* Bnt) {
  const int lane = threadIdx.x & 31;
  const bf16* p  = Bnt + lane * BF_LANE;
  return frag_from(p, p + 8);
}

// ============================================================================
// Kernel 1: LayerNorm statistics over (S,F) per (b,c). 64 blocks x 256 thr.
// ============================================================================
__global__ __launch_bounds__(256) void k_stats(const float* __restrict__ x,
                                               float* __restrict__ stats) {
  const int bc = blockIdx.x, tid = threadIdx.x;
  const int lane = tid & 31, wave = tid >> 5;
  const float4* p = (const float4*)(x + (size_t)bc * (S_ * F_));
  float s = 0.f, ss = 0.f;
  for (int i = tid; i < (S_ * F_ / 4); i += 256) {
    float4 v = p[i];
    s  += v.x + v.y + v.z + v.w;
    ss += v.x * v.x + v.y * v.y + v.z * v.z + v.w * v.w;
  }
#pragma unroll
  for (int off = 16; off > 0; off >>= 1) {
    s  += __shfl_down(s, off, 32);
    ss += __shfl_down(ss, off, 32);
  }
  __shared__ float sb[8], ssb[8];
  if (lane == 0) { sb[wave] = s; ssb[wave] = ss; }
  __syncthreads();
  if (tid == 0) {
    float S = 0.f, SS = 0.f;
#pragma unroll
    for (int i = 0; i < 8; ++i) { S += sb[i]; SS += ssb[i]; }
    const float n = (float)(S_ * F_);
    const float mean = S / n;
    const float var  = SS / n - mean * mean;
    stats[bc]      = mean;
    stats[64 + bc] = rsqrtf(var + 1e-5f);
  }
}

// ============================================================================
// Kernel 2: attention scalars q[b,s,o], k[b,s,c].  grid (S,B), 256 thr.
// ============================================================================
__global__ __launch_bounds__(256) void k_aux(
    const float* __restrict__ x, const float* __restrict__ stats,
    const float* __restrict__ pb3, const float* __restrict__ q1,
    const float* __restrict__ q2, const float* __restrict__ key,
    const float* __restrict__ qw, const float* __restrict__ kw,
    float* __restrict__ qbuf, float* __restrict__ kbuf) {
  const int s = blockIdx.x, b = blockIdx.y;
  const int tid = threadIdx.x, lane = tid & 31, wave = tid >> 5;
  const int c = wave, bc = b * C_ + c;
  const float mu = stats[bc], rstd = stats[64 + bc];

  const float4* xr  = (const float4*)(x   + ((size_t)bc * S_ + s) * F_);
  const float4* pr  = (const float4*)(pb3 + (size_t)s * F_);
  const float4* q1r = (const float4*)(q1  + ((size_t)s * C_ + c) * F_);
  const float4* kr  = (const float4*)(key + ((size_t)s * C_ + c) * F_);

  float qd = 0.f, kd = 0.f;
  for (int i = lane; i < F_ / 4; i += 32) {
    float4 xv = xr[i], pv = pr[i], qv = q1r[i], kv = kr[i];
    float a0 = ((xv.x - mu) * rstd + pv.x) * ND_INV;
    float a1 = ((xv.y - mu) * rstd + pv.y) * ND_INV;
    float a2 = ((xv.z - mu) * rstd + pv.z) * ND_INV;
    float a3 = ((xv.w - mu) * rstd + pv.w) * ND_INV;
    qd += a0 * qv.x + a1 * qv.y + a2 * qv.z + a3 * qv.w;
    kd += a0 * kv.x + a1 * kv.y + a2 * kv.z + a3 * kv.w;
  }
#pragma unroll
  for (int off = 16; off > 0; off >>= 1) {
    qd += __shfl_down(qd, off, 32);
    kd += __shfl_down(kd, off, 32);
  }
  __shared__ float sQ[C_], sK[C_];
  if (lane == 0) { sQ[c] = qd; sK[c] = kd; }
  __syncthreads();

  if (wave == 0 && lane < O_) {
    const int o = lane;
    float t = 0.f;
#pragma unroll
    for (int cc = 0; cc < C_; ++cc) t += q2[((size_t)s * O_ + o) * C_ + cc] * sQ[cc];
    float m = t;
#pragma unroll
    for (int off = 8; off > 0; off >>= 1) m = fmaxf(m, __shfl_xor(m, off, 32));
    float e = __expf(t - m);
    float sum = e;
#pragma unroll
    for (int off = 8; off > 0; off >>= 1) sum += __shfl_xor(sum, off, 32);
    qbuf[((size_t)b * S_ + s) * O_ + o] = (e / sum) * qw[(size_t)s * O_ + o];
  }
  if (wave == 1 && lane < C_) {
    const int cc = lane;
    float t = sK[cc], m = t;
#pragma unroll
    for (int off = 4; off > 0; off >>= 1) m = fmaxf(m, __shfl_xor(m, off, 32));
    float e = __expf(t - m);
    float sum = e;
#pragma unroll
    for (int off = 4; off > 0; off >>= 1) sum += __shfl_xor(sum, off, 32);
    kbuf[((size_t)b * S_ + s) * C_ + cc] = (e / sum) * kw[(size_t)s * C_ + cc];
  }
}

// ============================================================================
// Kernel 3: val(b,c,s,e) = relu( ((xn+pb2)/ND) @ value[c] )
// block tile 128(S) x 128(E), K-chunks 32, double-buffered. grid (4,4,64).
// ============================================================================
#define A3_STR 40
__global__ __launch_bounds__(256) void k_value_gemm(
    const float* __restrict__ x, const float* __restrict__ stats,
    const float* __restrict__ pb2, const float* __restrict__ value,
    float* __restrict__ val) {
  const int et = blockIdx.x, st = blockIdx.y, bc = blockIdx.z;
  const int c  = bc & (C_ - 1);
  const int tid = threadIdx.x, lane = tid & 31, wave = tid >> 5;
  const int s0 = st * 128, e0 = et * 128;
  const float mu = stats[bc], rstd = stats[64 + bc];

  __shared__ bf16 Ash[2][128 * A3_STR];   // row-major, LN fused
  __shared__ bf16 Bsh[2][8 * BF_NT];      // fragment-major

  v8f acc[8];
#pragma unroll
  for (int i = 0; i < 8; ++i)
#pragma unroll
    for (int j = 0; j < 8; ++j) acc[i][j] = 0.f;

  const float* xb = x     + (size_t)bc * (S_ * F_);
  const float* vb = value + (size_t)c  * (F_ * E_);

  auto stage = [&](int kt, int buf) {
    const int k0 = kt * 32;
    { // A tile: 128 rows x 32 K
      const int r = tid >> 1, f0 = (tid & 1) * 16;
      const float4* src = (const float4*)(xb  + (size_t)(s0 + r) * F_ + k0 + f0);
      const float4* pbp = (const float4*)(pb2 + (size_t)(s0 + r) * F_ + k0 + f0);
      bf16* dst = Ash[buf] + r * A3_STR + f0;
#pragma unroll
      for (int i = 0; i < 4; ++i) {
        float4 v = src[i], p = pbp[i];
        v4bf t = { (bf16)(((v.x - mu) * rstd + p.x) * ND_INV),
                   (bf16)(((v.y - mu) * rstd + p.y) * ND_INV),
                   (bf16)(((v.z - mu) * rstd + p.z) * ND_INV),
                   (bf16)(((v.w - mu) * rstd + p.w) * ND_INV) };
        *(v4bf*)(dst + i * 4) = t;   // ds_store_b64
      }
    }
    { // B tile: 32 K x 128 E, fragment-major, K-pair packed
      const int k  = (tid >> 4) * 2;
      const int eb = (tid & 15) * 8;
      const float4* r0 = (const float4*)(vb + (size_t)(k0 + k)     * E_ + e0 + eb);
      const float4* r1 = (const float4*)(vb + (size_t)(k0 + k + 1) * E_ + e0 + eb);
      float4 a0 = r0[0], a1 = r0[1], b0 = r1[0], b1 = r1[1];
      float r0v[8] = { a0.x, a0.y, a0.z, a0.w, a1.x, a1.y, a1.z, a1.w };
      float r1v[8] = { b0.x, b0.y, b0.z, b0.w, b1.x, b1.y, b1.z, b1.w };
      const int half = (k >= 16) ? 16 : 0;
      const int idx  = k & 15;
#pragma unroll
      for (int j = 0; j < 8; ++j) {
        const int e = eb + j;
        bf16* d = Bsh[buf] + (e >> 4) * BF_NT + (half + (e & 15)) * BF_LANE + idx;
        v2bf t = { (bf16)r0v[j], (bf16)r1v[j] };
        *(v2bf*)d = t;               // ds_store_b32
      }
    }
  };

  stage(0, 0);
  for (int kt = 0; kt < F_ / 32; ++kt) {
    const int buf = kt & 1;
    __syncthreads();                       // staged tiles ready / prev reads done
    if (kt + 1 < F_ / 32) stage(kt + 1, buf ^ 1);   // overlap with WMMA below
    v16bf a = load_a_frag(Ash[buf] + wave * 16 * A3_STR, A3_STR);
#pragma unroll
    for (int nt = 0; nt < 8; ++nt)
      acc[nt] = wmma_bf16(a, load_b_frag(Bsh[buf] + nt * BF_NT), acc[nt]);
  }

  float* ob = val + (size_t)bc * (S_ * E_);
  const int n = lane & 15, rb = (lane >> 4) * 8;
#pragma unroll
  for (int nt = 0; nt < 8; ++nt)
#pragma unroll
    for (int j = 0; j < 8; ++j)
      ob[(size_t)(s0 + wave * 16 + rb + j) * E_ + e0 + nt * 16 + n] =
          fmaxf(acc[nt][j], 0.f);
}

// ============================================================================
// Kernel 4: out = relu( (ls[s] @ (xn+pb1)/ND) @ rs[s] ) + q ⊗ (k · val)
// grid (2,S), 256 thr = 8 waves (wave == batch b), 16 N-tiles per wave.
// Dynamic LDS: Ash 128x520 bf16 | region2 { Xsh 64x512 / 2x Bsh + KV + Q }
// ============================================================================
#define A4_STR 520
#define B4_BUF (16 * BF_NT)                 // one B buffer, bf16 elements
#define SMEM4_BYTES (128 * A4_STR * 2 + 64 * 512 * 2)
__global__ __launch_bounds__(256) void k_final(
    const float* __restrict__ x, const float* __restrict__ stats,
    const float* __restrict__ pb1, const float* __restrict__ ls,
    const float* __restrict__ rs, const float* __restrict__ val,
    const float* __restrict__ qbuf, const float* __restrict__ kbuf,
    float* __restrict__ out) {
  const int eh = blockIdx.x, s = blockIdx.y;
  const int eoff = eh * 256;
  const int tid = threadIdx.x, lane = tid & 31, wave = tid >> 5;

  extern __shared__ char smem[];
  bf16*  Ash  = (bf16*)smem;                         // 128 x A4_STR
  char*  r2   = smem + 128 * A4_STR * 2;
  bf16*  Xsh  = (bf16*)r2;                           // 64 x 512 (phases 0-1)
  bf16*  Bsh  = (bf16*)r2;                           // 2 x B4_BUF (phase 3)
  float* KVsh = (float*)(r2 + 2 * B4_BUF * 2);       // 8 x 256
  float* Qsh  = KVsh + 8 * 256;                      // 128

  // ---- phase 0: Xsh[b*8+c][f] = ((x - mu)*rstd + pb1)/ND ----
  {
    const int r = tid >> 2, f0 = (tid & 3) * 128;
    const float mu = stats[r], rstd = stats[64 + r];
    const float4* src = (const float4*)(x   + ((size_t)r * S_ + s) * F_ + f0);
    const float4* pbp = (const float4*)(pb1 + (size_t)s * F_ + f0);
    bf16* dst = Xsh + r * 512 + f0;
#pragma unroll 8
    for (int i = 0; i < 32; ++i) {
      float4 v = src[i], p = pbp[i];
      v4bf t = { (bf16)(((v.x - mu) * rstd + p.x) * ND_INV),
                 (bf16)(((v.y - mu) * rstd + p.y) * ND_INV),
                 (bf16)(((v.z - mu) * rstd + p.z) * ND_INV),
                 (bf16)(((v.w - mu) * rstd + p.w) * ND_INV) };
      *(v4bf*)(dst + i * 4) = t;
    }
  }
  __syncthreads();

  // ---- phase 1: Ash[b*16+o][f] = sum_c ls[s,o,c] * Xsh[b*8+c][f] ----
  {
    const int r = tid >> 1, f0 = (tid & 1) * 256;
    const int b = r >> 4, o = r & 15;
    float w[C_];
#pragma unroll
    for (int cc = 0; cc < C_; ++cc) w[cc] = ls[((size_t)s * O_ + o) * C_ + cc];
    const bf16* xrow = Xsh + (b * C_) * 512 + f0;
    bf16* dst = Ash + r * A4_STR + f0;
    for (int f = 0; f < 256; f += 4) {
      float a0 = 0.f, a1 = 0.f, a2 = 0.f, a3 = 0.f;
#pragma unroll
      for (int cc = 0; cc < C_; ++cc) {
        const bf16* xc = xrow + cc * 512 + f;
        a0 += w[cc] * (float)xc[0];
        a1 += w[cc] * (float)xc[1];
        a2 += w[cc] * (float)xc[2];
        a3 += w[cc] * (float)xc[3];
      }
      v4bf t = { (bf16)a0, (bf16)a1, (bf16)a2, (bf16)a3 };
      *(v4bf*)(dst + f) = t;
    }
  }
  __syncthreads();   // last read of Xsh before region2 reuse

  // ---- phase 2: KVsh[b][e'] = sum_c k[b,s,c]*val[b,c,s,e] ; Qsh ----
  {
    const int b = tid >> 5, j = tid & 31, ep = j * 8;
    float kc[C_];
#pragma unroll
    for (int cc = 0; cc < C_; ++cc) kc[cc] = kbuf[((size_t)b * S_ + s) * C_ + cc];
    float kv[8];
#pragma unroll
    for (int i = 0; i < 8; ++i) kv[i] = 0.f;
#pragma unroll
    for (int cc = 0; cc < C_; ++cc) {
      const float* vp = val + ((size_t)(b * C_ + cc) * S_ + s) * E_ + eoff + ep;
      float4 v0 = *(const float4*)vp;
      float4 v1 = *(const float4*)(vp + 4);
      kv[0] += kc[cc] * v0.x; kv[1] += kc[cc] * v0.y;
      kv[2] += kc[cc] * v0.z; kv[3] += kc[cc] * v0.w;
      kv[4] += kc[cc] * v1.x; kv[5] += kc[cc] * v1.y;
      kv[6] += kc[cc] * v1.z; kv[7] += kc[cc] * v1.w;
    }
#pragma unroll
    for (int i = 0; i < 8; ++i) KVsh[b * 256 + ep + i] = kv[i];
    if (tid < 128) Qsh[tid] = qbuf[((size_t)(tid >> 4) * S_ + s) * O_ + (tid & 15)];
  }

  // ---- phase 3: (128 x 512) @ (512 x 256), double-buffered B stream ----
  v8f acc[16];
#pragma unroll
  for (int i = 0; i < 16; ++i)
#pragma unroll
    for (int j = 0; j < 8; ++j) acc[i][j] = 0.f;

  const float* rsb = rs + (size_t)s * F_ * E_;

  auto stageB = [&](int kt, int buf) {
    const int k0 = kt * 32;
    const int k  = (tid >> 4) * 2;
    const int eb = (tid & 15) * 16;      // one nt group per thread
    const int nt = tid & 15;
    const float4* r0 = (const float4*)(rsb + (size_t)(k0 + k)     * E_ + eoff + eb);
    const float4* r1 = (const float4*)(rsb + (size_t)(k0 + k + 1) * E_ + eoff + eb);
    float4 a0 = r0[0], a1 = r0[1], a2 = r0[2], a3 = r0[3];
    float4 b0 = r1[0], b1 = r1[1], b2 = r1[2], b3 = r1[3];
    float r0v[16] = { a0.x,a0.y,a0.z,a0.w, a1.x,a1.y,a1.z,a1.w,
                      a2.x,a2.y,a2.z,a2.w, a3.x,a3.y,a3.z,a3.w };
    float r1v[16] = { b0.x,b0.y,b0.z,b0.w, b1.x,b1.y,b1.z,b1.w,
                      b2.x,b2.y,b2.z,b2.w, b3.x,b3.y,b3.z,b3.w };
    const int half = (k >= 16) ? 16 : 0;
    const int idx  = k & 15;
    bf16* dnt = Bsh + buf * B4_BUF + nt * BF_NT + idx;
#pragma unroll
    for (int j = 0; j < 16; ++j) {
      v2bf t = { (bf16)r0v[j], (bf16)r1v[j] };
      *(v2bf*)(dnt + (half + j) * BF_LANE) = t;   // ds_store_b32
    }
    if (kt + 1 < F_ / 32)
      __builtin_prefetch(rsb + (size_t)(k0 + 32 + k) * E_ + eoff + eb, 0, 0);
  };

  stageB(0, 0);
  for (int kt = 0; kt < F_ / 32; ++kt) {
    const int buf = kt & 1;
    __syncthreads();                     // buf staged / buf^1 reads retired
    if (kt + 1 < F_ / 32) stageB(kt + 1, buf ^ 1);
    const int k0 = kt * 32;
    v16bf a = load_a_frag(Ash + (wave * 16) * A4_STR + k0, A4_STR);
#pragma unroll
    for (int nt = 0; nt < 16; ++nt)
      acc[nt] = wmma_bf16(a, load_b_frag(Bsh + buf * B4_BUF + nt * BF_NT), acc[nt]);
  }
  __syncthreads();

  // ---- phase 4: out = relu(acc) + q[b,s,o] * kv[b,e] ----
  const int b = wave, n = lane & 15, rb = (lane >> 4) * 8;
#pragma unroll
  for (int nt = 0; nt < 16; ++nt)
#pragma unroll
    for (int j = 0; j < 8; ++j) {
      const int o = rb + j, col = nt * 16 + n;
      const float v = fmaxf(acc[nt][j], 0.f) + Qsh[b * O_ + o] * KVsh[b * 256 + col];
      out[((size_t)(b * S_ + s) * O_ + o) * E_ + eoff + col] = v;
    }
}

// ============================================================================
// Launch
// ============================================================================
extern "C" void kernel_launch(void* const* d_in, const int* in_sizes, int n_in,
                              void* d_out, int out_size, void* d_ws, size_t ws_size,
                              hipStream_t stream) {
  (void)in_sizes; (void)n_in; (void)out_size; (void)ws_size;
  const float* x     = (const float*)d_in[0];
  const float* pb1   = (const float*)d_in[1];
  const float* pb2   = (const float*)d_in[2];
  const float* pb3   = (const float*)d_in[3];
  const float* q1    = (const float*)d_in[4];
  const float* q2    = (const float*)d_in[5];
  const float* key   = (const float*)d_in[6];
  const float* val_w = (const float*)d_in[7];
  const float* qw    = (const float*)d_in[8];
  const float* kw    = (const float*)d_in[9];
  const float* ls    = (const float*)d_in[10];
  const float* rs    = (const float*)d_in[11];
  float* out = (float*)d_out;

  float* stats = (float*)d_ws;
  float* qbuf  = stats + 128;
  float* kbuf  = qbuf + (size_t)B_ * S_ * O_;
  float* val   = kbuf + (size_t)B_ * S_ * C_;

  k_stats<<<B_ * C_, 256, 0, stream>>>(x, stats);
  k_aux<<<dim3(S_, B_), 256, 0, stream>>>(x, stats, pb3, q1, q2, key, qw, kw,
                                          qbuf, kbuf);
  k_value_gemm<<<dim3(E_ / 128, S_ / 128, B_ * C_), 256, 0, stream>>>(
      x, stats, pb2, val_w, val);
  k_final<<<dim3(2, S_), 256, SMEM4_BYTES, stream>>>(
      x, stats, pb1, ls, rs, val, qbuf, kbuf, out);
}
